// LinearE2V_10290741641936
// MI455X (gfx1250) — matrix operand order repaired and checked by reading, not compiled, not run on hardware
//
#include <hip/hip_runtime.h>
#include <hip/hip_bf16.h>
#include <stdint.h>

#define N_V 8192
#define N_E 16384
#define DIM 128
#define HID 256
#define KSPLIT 4
#define KCHUNK (N_E / KSPLIT)

typedef __bf16 bf16;
typedef __attribute__((ext_vector_type(16))) __bf16 v16bf;
typedef __attribute__((ext_vector_type(8)))  float  v8f;
typedef __attribute__((ext_vector_type(4)))  float  vf4;
typedef __attribute__((ext_vector_type(4)))  unsigned int v4u;
typedef __attribute__((ext_vector_type(8)))  int    v8i;
typedef __attribute__((ext_vector_type(4)))  int    v4i;

#if defined(__has_builtin)
#if __has_builtin(__builtin_amdgcn_tensor_load_to_lds)
#define HAVE_TDM 1
#endif
#endif

// ---------------- WMMA fragment helpers (CDNA5 wave32 layouts) ----------------

// A-matrix 16x32 bf16 from row-major LDS tile (row stride ld elements).
// lane<16: row=lane, K = {0..7, 16..23}; lane>=16: row=lane-16, K = {8..15, 24..31}
__device__ inline v16bf load_a_frag_lds(const bf16* p, int ld) {
  int lane = threadIdx.x & 31;
  const bf16* row = p + (lane & 15) * ld + ((lane >> 4) << 3);
  v16bf a;
#pragma unroll
  for (int i = 0; i < 8; ++i) a[i] = row[i];
#pragma unroll
  for (int i = 0; i < 8; ++i) a[8 + i] = row[16 + i];
  return a;
}

// Same A fragment but from an fp32 row-major LDS tile (ld=128), converting to bf16.
__device__ inline v16bf load_a_frag_f32_lds(const float* p, int koff) {
  int lane = threadIdx.x & 31;
  const float* row = p + (lane & 15) * 128 + koff + ((lane >> 4) << 3);
  vf4 f0 = *(const vf4*)(row);
  vf4 f1 = *(const vf4*)(row + 4);
  vf4 f2 = *(const vf4*)(row + 16);
  vf4 f3 = *(const vf4*)(row + 20);
  v16bf a;
#pragma unroll
  for (int i = 0; i < 4; ++i) {
    a[i]      = (bf16)f0[i];
    a[4 + i]  = (bf16)f1[i];
    a[8 + i]  = (bf16)f2[i];
    a[12 + i] = (bf16)f3[i];
  }
  return a;
}

// B-matrix 32x16 bf16 from our pre-packed fragment buffer:
// frag f stores, per lane, 16 contiguous halves (K=32kt+{8h..8h+7, 16+8h..23+8h} of column n, h=lane/16).
__device__ inline v16bf load_b_frag(const bf16* pack, int frag) {
  int lane = threadIdx.x & 31;
  const bf16* p = pack + ((size_t)frag * 32 + lane) * 16;
  v16bf b;
#pragma unroll
  for (int i = 0; i < 16; ++i) b[i] = p[i];
  return b;
}

__device__ inline v8f wmma_bf16(v16bf a, v16bf b, v8f c) {
  return __builtin_amdgcn_wmma_f32_16x16x32_bf16(false, a, false, b, (short)0, c,
                                                 false, false);
}

#ifdef HAVE_TDM
// Issue one TDM load of a 16-row x 128-col fp32 tile (row stride 16384 elements)
// from global memory into LDS at lds_addr.  D# per CDNA5 ISA ch.8.
// This toolchain exposes the 6-arg builtin:
//   (uint32x4 g0, int32x8 g1, int32x4 g2, int32x4 g3, int32x8, i32 cpol)
__device__ inline void tdm_load_tile(uint32_t lds_addr, const float* gptr) {
  uint64_t ga = (uint64_t)(uintptr_t)gptr;
  v4u g0;
  g0[0] = 1u;                                   // count=1 (valid), load, no gather
  g0[1] = lds_addr;                             // LDS byte address
  g0[2] = (uint32_t)ga;                         // global_addr[31:0]
  g0[3] = (uint32_t)((ga >> 32) & 0x01ffffffu)  // global_addr[56:32]
          | (2u << 30);                         // type=2 ("image")
  v8i g1;
  g1[0] = (int)(2u << 16);        // workgroup_mask=0, data_size=2 (4 bytes)
  g1[1] = (int)(16384u << 16);    // tensor_dim0[15:0]  (=16384 elements)
  g1[2] = (int)(16384u << 16);    // tensor_dim0[31:16]=0 | tensor_dim1[15:0]=16384
  g1[3] = (int)(128u << 16);      // tensor_dim1[31:16]=0 | tile_dim0=128
  g1[4] = 16;                     // tile_dim1=16, tile_dim2=0
  g1[5] = 16384;                  // tensor_dim0_stride[31:0] (elements)
  g1[6] = 0;                      // stride hi / tensor_dim1_stride lo
  g1[7] = 0;
  v4i z4 = {0, 0, 0, 0};
  v8i z8 = {0, 0, 0, 0, 0, 0, 0, 0};
  __builtin_amdgcn_tensor_load_to_lds(g0, g1, z4, z4, z8, 0);
}
#endif

// ---------------- per-row layernorm stats (wave32 reduce) ----------------
__device__ inline void row_stats(const float x[4], float& mu, float& rs) {
  float s = x[0] + x[1] + x[2] + x[3];
  float q = x[0]*x[0] + x[1]*x[1] + x[2]*x[2] + x[3]*x[3];
#pragma unroll
  for (int off = 16; off > 0; off >>= 1) {
    s += __shfl_xor(s, off, 32);
    q += __shfl_xor(q, off, 32);
  }
  mu = s * (1.f / 128.f);
  float var = q * (1.f / 128.f) - mu * mu;
  rs = rsqrtf(var + 1e-5f);
}

// ---------------- 16-row MLP tile: relu(A@W1+b1)@W2+b2, KTILES = K/32 ----------------
template <int KTILES>
__device__ inline void mlp_tile(const bf16* __restrict__ A,      // LDS [16][KTILES*32]
                                const bf16* __restrict__ w1p, const float* __restrict__ b1,
                                const bf16* __restrict__ w2p, const float* __restrict__ b2,
                                bf16* __restrict__ Hidb,         // LDS [16][256]
                                v8f (&out)[8]) {
  int lane = threadIdx.x & 31;
  int nlo = lane & 15;
  int mhi = (lane >> 4) << 3;

  v16bf afr[KTILES];
#pragma unroll
  for (int kt = 0; kt < KTILES; ++kt)
    afr[kt] = load_a_frag_lds(A + kt * 32, KTILES * 32);

  // GEMM1: [16 x K] @ [K x 256] + b1, ReLU -> Hid (bf16 LDS)
  for (int nt = 0; nt < 16; ++nt) {
    float bv = b1[nt * 16 + nlo];
    v8f acc;
#pragma unroll
    for (int v = 0; v < 8; ++v) acc[v] = bv;
#pragma unroll
    for (int kt = 0; kt < KTILES; ++kt) {
      v16bf b = load_b_frag(w1p, kt * 16 + nt);
      acc = wmma_bf16(afr[kt], b, acc);
    }
#pragma unroll
    for (int v = 0; v < 8; ++v) {
      float f = acc[v] > 0.f ? acc[v] : 0.f;
      Hidb[(mhi + v) * 256 + nt * 16 + nlo] = (bf16)f;
    }
  }

  v16bf hfr[8];
#pragma unroll
  for (int kt = 0; kt < 8; ++kt)
    hfr[kt] = load_a_frag_lds(Hidb + kt * 32, 256);

  // GEMM2: [16 x 256] @ [256 x 128] + b2
  for (int nt = 0; nt < 8; ++nt) {
    float bv = b2[nt * 16 + nlo];
    v8f acc;
#pragma unroll
    for (int v = 0; v < 8; ++v) acc[v] = bv;
#pragma unroll
    for (int kt = 0; kt < 8; ++kt) {
      v16bf b = load_b_frag(w2p, kt * 8 + nt);
      acc = wmma_bf16(hfr[kt], b, acc);
    }
    out[nt] = acc;
  }
}

// ---------------- kernels ----------------

// PE tables + zero colsum
__global__ void k_pe(float* __restrict__ pe1, float* __restrict__ pe2,
                     float* __restrict__ colsum) {
  int i = blockIdx.x * blockDim.x + threadIdx.x;
  if (i < DIM) colsum[i] = 0.f;
  const float c = -9.210340371976184f / 128.f;  // -ln(10000)/D
  if (i < 9 * DIM) {
    int k = i >> 7, d = i & 127;
    float ang = (float)k * __expf((float)(d & ~1) * c);
    pe1[i] = (d & 1) ? __cosf(ang) : __sinf(ang);
  } else if (i < 11 * DIM) {
    int j = i - 9 * DIM;
    int k = j >> 7, d = j & 127;
    float ang = (float)k * __expf((float)(d & ~1) * c);
    pe2[j] = (d & 1) ? __cosf(ang) : __sinf(ang);
  }
}

// Pack fp32 row-major [Kin][Nout] into bf16 WMMA B-fragment order.
__global__ void k_pack(const float* __restrict__ src, bf16* __restrict__ dst,
                       int Kin, int Nout) {
  int idx = blockIdx.x * blockDim.x + threadIdx.x;
  if (idx >= Kin * Nout) return;
  int Ntiles = Nout >> 4;
  int f = idx >> 9;         // 512 halves / frag
  int rem = idx & 511;
  int lane = rem >> 4;
  int i = rem & 15;
  int kt = f / Ntiles, nt = f - kt * Ntiles;
  int n = (nt << 4) + (lane & 15);
  int k = (kt << 5) + ((lane >> 4) << 3) + (i & 7) + ((i >> 3) << 4);
  dst[idx] = (bf16)src[(size_t)k * Nout + n];
}

// Stage 1: LN + MLP1 + residual for x_v (pe1[1]) and x_e (pe1[edge_orders]).
__global__ __launch_bounds__(64) void k_stage1(
    const float* __restrict__ xv, const float* __restrict__ xe,
    const int* __restrict__ eord,
    const float* __restrict__ g1, const float* __restrict__ bn1,
    const float* __restrict__ pe1,
    const bf16* __restrict__ w1p, const float* __restrict__ b1,
    const bf16* __restrict__ w2p, const float* __restrict__ b2,
    float* __restrict__ xv1, float* __restrict__ xe1,
    float* __restrict__ colsum) {
  __shared__ __align__(16) bf16 Abuf[2][16 * 256];
  __shared__ __align__(16) bf16 Hbuf[2][16 * 256];
  int wave = threadIdx.x >> 5, lane = threadIdx.x & 31;
  int tile = blockIdx.x * 2 + wave;
  const int NVT = N_V / 16;
  bool isV = tile < NVT;
  const float* X = isV ? xv : xe;
  int r0 = (isV ? tile : tile - NVT) * 16;
  bf16* A = Abuf[wave];
  bf16* H = Hbuf[wave];
  int c = lane * 4;

  for (int r = 0; r < 16; ++r) {
    const float* row = X + (size_t)(r0 + r) * DIM;
    vf4 xf = *(const vf4*)(row + c);
    float xr[4] = {xf[0], xf[1], xf[2], xf[3]};
    float mu, rs;
    row_stats(xr, mu, rs);
    int pidx = isV ? 1 : eord[r0 + r];
    const float* per = pe1 + pidx * DIM;
#pragma unroll
    for (int j = 0; j < 4; ++j) {
      float t = (xr[j] - mu) * rs * g1[c + j] + bn1[c + j];
      A[r * 256 + c + j] = (bf16)t;
      A[r * 256 + 128 + c + j] = (bf16)per[c + j];
    }
  }

  v8f out[8];
  mlp_tile<8>(A, w1p, b1, w2p, b2, H, out);

  float* Y = (isV ? xv1 : xe1) + (size_t)r0 * DIM;
  const float* Xr = X + (size_t)r0 * DIM;
  int nlo = lane & 15, mhi = (lane >> 4) << 3;
#pragma unroll
  for (int nt = 0; nt < 8; ++nt) {
    float ps = 0.f;
#pragma unroll
    for (int v = 0; v < 8; ++v) {
      int m = mhi + v, n = nt * 16 + nlo;
      float f = out[nt][v] + Xr[m * DIM + n];
      Y[m * DIM + n] = f;
      ps += f;
    }
    atomicAdd(&colsum[nt * 16 + nlo], ps);  // for x0 = mean(...)
  }
}

// Big GEMM: partial[kc] = incidence[:, kc-slab] @ x_e'
// TDM double-buffered staging of the 512MB incidence stream into LDS,
// fp32->bf16 in-register, v_wmma_f32_16x16x32_bf16 accumulate.
__global__ __launch_bounds__(64) void k_inc(const float* __restrict__ inc,
                                            const bf16* __restrict__ xep,
                                            float* __restrict__ part) {
  int wave = __builtin_amdgcn_readfirstlane(threadIdx.x >> 5);
  int lane = threadIdx.x & 31;
  int job = blockIdx.x * 2 + wave;
  int rowtile = job >> 2;  // KSPLIT = 4
  int kc = job & 3;
  const float* A0 = inc + (size_t)rowtile * 16 * N_E + (size_t)kc * KCHUNK;

  v8f acc[8];
#pragma unroll
  for (int nt = 0; nt < 8; ++nt)
#pragma unroll
    for (int v = 0; v < 8; ++v) acc[nt][v] = 0.f;

#ifdef HAVE_TDM
  __shared__ __align__(16) float Sbuf[2][2][16 * 128];  // [wave][dbuf], 8KB tiles
  uint32_t ldsA[2];
  ldsA[0] = (uint32_t)(uintptr_t)(&Sbuf[wave][0][0]);
  ldsA[1] = (uint32_t)(uintptr_t)(&Sbuf[wave][1][0]);
  const int NTL = KCHUNK / 128;  // 32 staged tiles of 16x128 fp32

  tdm_load_tile(ldsA[0], A0);
  for (int t = 0; t < NTL; ++t) {
    if (t + 1 < NTL) {
      tdm_load_tile(ldsA[(t + 1) & 1], A0 + (t + 1) * 128);
      asm volatile("s_wait_tensorcnt 0x1" ::: "memory");  // tile t resident
    } else {
      asm volatile("s_wait_tensorcnt 0x0" ::: "memory");
    }
    const float* lb = &Sbuf[wave][t & 1][0];
#pragma unroll
    for (int k2 = 0; k2 < 4; ++k2) {
      v16bf a = load_a_frag_f32_lds(lb, k2 * 32);
      int ktg = kc * (KCHUNK / 32) + t * 4 + k2;
#pragma unroll
      for (int nt = 0; nt < 8; ++nt) {
        v16bf b = load_b_frag(xep, ktg * 8 + nt);
        acc[nt] = wmma_bf16(a, b, acc[nt]);
      }
    }
  }
#else
  int m = lane & 15;
  int klo = (lane >> 4) << 3;
  for (int kt = 0; kt < KCHUNK / 32; ++kt) {
    const float* row = A0 + (size_t)m * N_E + kt * 32 + klo;
    vf4 f0 = *(const vf4*)(row);
    vf4 f1 = *(const vf4*)(row + 4);
    vf4 f2 = *(const vf4*)(row + 16);
    vf4 f3 = *(const vf4*)(row + 20);
    v16bf a;
#pragma unroll
    for (int i = 0; i < 4; ++i) {
      a[i]      = (bf16)f0[i];
      a[4 + i]  = (bf16)f1[i];
      a[8 + i]  = (bf16)f2[i];
      a[12 + i] = (bf16)f3[i];
    }
    int ktg = kc * (KCHUNK / 32) + kt;
#pragma unroll
    for (int nt = 0; nt < 8; ++nt) {
      v16bf b = load_b_frag(xep, ktg * 8 + nt);
      acc[nt] = wmma_bf16(a, b, acc[nt]);
    }
  }
#endif

  int r0 = rowtile * 16;
  int nlo = lane & 15, mhi = (lane >> 4) << 3;
  float* out = part + (size_t)kc * N_V * DIM;
#pragma unroll
  for (int nt = 0; nt < 8; ++nt)
#pragma unroll
    for (int v = 0; v < 8; ++v)
      out[(size_t)(r0 + mhi + v) * DIM + nt * 16 + nlo] = acc[nt][v];
}

// x0 path: mean -> LN -> MLP2 (scalar, 1 row) -> residual
__global__ void k_x0(const float* __restrict__ colsum, const float* __restrict__ pe2,
                     const float* __restrict__ g, const float* __restrict__ bn,
                     const float* __restrict__ w1, const float* __restrict__ b1,
                     const float* __restrict__ w2, const float* __restrict__ b2,
                     float* __restrict__ x0) {
  __shared__ float a[2 * DIM];
  __shared__ float hid[HID];
  __shared__ float red[DIM];
  __shared__ float red2[DIM];
  __shared__ float stats[2];
  int t = threadIdx.x;  // 128 threads
  float xv = colsum[t] * (1.f / (float)(N_V + N_E));
  red[t] = xv;
  red2[t] = xv * xv;
  __syncthreads();
  for (int s = 64; s > 0; s >>= 1) {
    if (t < s) { red[t] += red[t + s]; red2[t] += red2[t + s]; }
    __syncthreads();
  }
  if (t == 0) {
    float mu = red[0] * (1.f / 128.f);
    float var = red2[0] * (1.f / 128.f) - mu * mu;
    stats[0] = mu;
    stats[1] = rsqrtf(var + 1e-5f);
  }
  __syncthreads();
  a[t] = (xv - stats[0]) * stats[1] * g[t] + bn[t];
  a[DIM + t] = pe2[t];  // pe2 row 0
  __syncthreads();
  for (int h = t; h < HID; h += 128) {
    float accv = b1[h];
    for (int k = 0; k < 2 * DIM; ++k) accv += a[k] * w1[(size_t)k * HID + h];
    hid[h] = accv > 0.f ? accv : 0.f;
  }
  __syncthreads();
  float o = b2[t];
  for (int h = 0; h < HID; ++h) o += hid[h] * w2[(size_t)h * DIM + t];
  x0[t] = xv + o;
}

// Stage 2+3 fused: x1 reduce/scale, LN+MLP2+res, add x0, LN+MLP3+res, +bias
__global__ __launch_bounds__(64) void k_stage23(
    const float* __restrict__ xv1, const float* __restrict__ part,
    const float* __restrict__ suffix, const float* __restrict__ x0,
    const float* __restrict__ pe2,
    const float* __restrict__ g2, const float* __restrict__ bn2,
    const float* __restrict__ g3, const float* __restrict__ bn3,
    const bf16* __restrict__ w21p, const float* __restrict__ b21,
    const bf16* __restrict__ w22p, const float* __restrict__ b22,
    const bf16* __restrict__ w31p, const float* __restrict__ b31,
    const bf16* __restrict__ w32p, const float* __restrict__ b32,
    const float* __restrict__ bias_b, float* __restrict__ outp) {
  __shared__ __align__(16) float Xbuf[2][16 * DIM];
  __shared__ __align__(16) bf16 Abuf[2][16 * 256];
  __shared__ __align__(16) bf16 Hbuf[2][16 * 256];
  int wave = threadIdx.x >> 5, lane = threadIdx.x & 31;
  int tile = blockIdx.x * 2 + wave;
  int r0 = tile * 16;
  float* Xs = Xbuf[wave];
  bf16* A = Abuf[wave];
  bf16* H = Hbuf[wave];
  int c = lane * 4;
  int nlo = lane & 15, mhi = (lane >> 4) << 3;

  // x1 = (x_v' + inc@x_e') / (1+suffix); stash fp32; LN -> A with pe2 row 1
  for (int r = 0; r < 16; ++r) {
    size_t idx = (size_t)(r0 + r) * DIM + c;
    vf4 v = *(const vf4*)(xv1 + idx);
    float xr[4] = {v[0], v[1], v[2], v[3]};
#pragma unroll
    for (int kc = 0; kc < KSPLIT; ++kc) {
      vf4 p = *(const vf4*)(part + (size_t)kc * N_V * DIM + idx);
      xr[0] += p[0]; xr[1] += p[1]; xr[2] += p[2]; xr[3] += p[3];
    }
    float inv = 1.f / (1.f + suffix[r0 + r]);
#pragma unroll
    for (int j = 0; j < 4; ++j) xr[j] *= inv;
    float mu, rs;
    row_stats(xr, mu, rs);
#pragma unroll
    for (int j = 0; j < 4; ++j) {
      Xs[r * DIM + c + j] = xr[j];
      A[r * 256 + c + j] = (bf16)((xr[j] - mu) * rs * g2[c + j] + bn2[c + j]);
      A[r * 256 + 128 + c + j] = (bf16)pe2[DIM + c + j];  // pe2 row 1
    }
  }

  v8f out[8];
  mlp_tile<8>(A, w21p, b21, w22p, b22, H, out);

  // x = x0 + x1 + mlp2out  (disjoint per-lane elements)
#pragma unroll
  for (int nt = 0; nt < 8; ++nt)
#pragma unroll
    for (int v = 0; v < 8; ++v) {
      int m = mhi + v, n = nt * 16 + nlo;
      Xs[m * DIM + n] = Xs[m * DIM + n] + out[nt][v] + x0[n];
    }

  // LN3(x) -> A2 (width 128, reuse A buffer)
  for (int r = 0; r < 16; ++r) {
    float xr[4] = {Xs[r * DIM + c], Xs[r * DIM + c + 1], Xs[r * DIM + c + 2],
                   Xs[r * DIM + c + 3]};
    float mu, rs;
    row_stats(xr, mu, rs);
#pragma unroll
    for (int j = 0; j < 4; ++j)
      A[r * DIM + c + j] = (bf16)((xr[j] - mu) * rs * g3[c + j] + bn3[c + j]);
  }

  v8f out2[8];
  mlp_tile<4>(A, w31p, b31, w32p, b32, H, out2);

#pragma unroll
  for (int nt = 0; nt < 8; ++nt)
#pragma unroll
    for (int v = 0; v < 8; ++v) {
      int m = mhi + v, n = nt * 16 + nlo;
      outp[(size_t)(r0 + m) * DIM + n] = Xs[m * DIM + n] + out2[nt][v] + bias_b[n];
    }
}

// ---------------- launch ----------------
extern "C" void kernel_launch(void* const* d_in, const int* in_sizes, int n_in,
                              void* d_out, int out_size, void* d_ws, size_t ws_size,
                              hipStream_t stream) {
  const float* x_v    = (const float*)d_in[0];
  const float* x_e    = (const float*)d_in[1];
  const float* inc    = (const float*)d_in[2];
  const float* suffix = (const float*)d_in[3];
  const int*   eord   = (const int*)d_in[4];
  const float* m1w1 = (const float*)d_in[5];  const float* m1b1 = (const float*)d_in[6];
  const float* m1w2 = (const float*)d_in[7];  const float* m1b2 = (const float*)d_in[8];
  const float* m2w1 = (const float*)d_in[9];  const float* m2b1 = (const float*)d_in[10];
  const float* m2w2 = (const float*)d_in[11]; const float* m2b2 = (const float*)d_in[12];
  const float* m3w1 = (const float*)d_in[13]; const float* m3b1 = (const float*)d_in[14];
  const float* m3w2 = (const float*)d_in[15]; const float* m3b2 = (const float*)d_in[16];
  const float* n1g = (const float*)d_in[17];  const float* n1b = (const float*)d_in[18];
  const float* n2g = (const float*)d_in[19];  const float* n2b = (const float*)d_in[20];
  const float* n3g = (const float*)d_in[21];  const float* n3b = (const float*)d_in[22];
  const float* bias = (const float*)d_in[23];
  float* outp = (float*)d_out;

  char* W = (char*)d_ws;
  size_t off = 0;
  auto carve = [&](size_t bytes) -> void* {
    void* p = W + off;
    off = (off + bytes + 255) & ~(size_t)255;
    return p;
  };
  float* pe1   = (float*)carve(9 * DIM * 4);
  float* pe2   = (float*)carve(2 * DIM * 4);
  bf16* pw11   = (bf16*)carve(256 * 256 * 2);
  bf16* pw12   = (bf16*)carve(256 * 128 * 2);
  bf16* pw21   = (bf16*)carve(256 * 256 * 2);
  bf16* pw22   = (bf16*)carve(256 * 128 * 2);
  bf16* pw31   = (bf16*)carve(128 * 256 * 2);
  bf16* pw32   = (bf16*)carve(256 * 128 * 2);
  bf16* xep    = (bf16*)carve((size_t)N_E * DIM * 2);
  float* xv1   = (float*)carve((size_t)N_V * DIM * 4);
  float* xe1   = (float*)carve((size_t)N_E * DIM * 4);
  float* part  = (float*)carve((size_t)KSPLIT * N_V * DIM * 4);
  float* csum  = (float*)carve(DIM * 4);
  float* x0b   = (float*)carve(DIM * 4);

  k_pe<<<6, 256, 0, stream>>>(pe1, pe2, csum);

  k_pack<<<(256 * 256 + 255) / 256, 256, 0, stream>>>(m1w1, pw11, 256, 256);
  k_pack<<<(256 * 128 + 255) / 256, 256, 0, stream>>>(m1w2, pw12, 256, 128);
  k_pack<<<(256 * 256 + 255) / 256, 256, 0, stream>>>(m2w1, pw21, 256, 256);
  k_pack<<<(256 * 128 + 255) / 256, 256, 0, stream>>>(m2w2, pw22, 256, 128);
  k_pack<<<(128 * 256 + 255) / 256, 256, 0, stream>>>(m3w1, pw31, 128, 256);
  k_pack<<<(256 * 128 + 255) / 256, 256, 0, stream>>>(m3w2, pw32, 256, 128);

  k_stage1<<<(N_V + N_E) / 32, 64, 0, stream>>>(x_v, x_e, eord, n1g, n1b, pe1,
                                                pw11, m1b1, pw12, m1b2,
                                                xv1, xe1, csum);

  k_pack<<<((size_t)N_E * DIM + 255) / 256, 256, 0, stream>>>(xe1, xep, N_E, DIM);

  k_inc<<<(N_V / 16) * KSPLIT / 2, 64, 0, stream>>>(inc, xep, part);

  k_x0<<<1, 128, 0, stream>>>(csum, pe2, n2g, n2b, m2w1, m2b1, m2w2, m2b2, x0b);

  k_stage23<<<N_V / 32, 64, 0, stream>>>(xv1, part, suffix, x0b, pe2,
                                         n2g, n2b, n3g, n3b,
                                         pw21, m2b1, pw22, m2b2,
                                         pw31, m3b1, pw32, m3b2,
                                         bias, outp);
}
